// MultiLayerGAT_45406394253682
// MI455X (gfx1250) — matrix operand order repaired and checked by reading, not compiled, not run on hardware
//
#include <hip/hip_runtime.h>

#define NNODES 50000
#define NEDGES 800000
#define NGRAPH 64
#define NEG_SLOPE 0.2f

#ifndef __has_builtin
#define __has_builtin(x) 0
#endif
#if __has_builtin(__builtin_amdgcn_global_load_async_to_lds_b128) && \
    __has_builtin(__builtin_amdgcn_s_wait_asynccnt)
#define HAVE_ASYNC_LDS 1
#else
#define HAVE_ASYNC_LDS 0
#endif

typedef __attribute__((ext_vector_type(16))) __bf16 v16bf;
typedef __attribute__((ext_vector_type(8)))  float  v8f;
typedef __attribute__((ext_vector_type(4)))  int    v4i;
#if HAVE_ASYNC_LDS
typedef __attribute__((address_space(1))) v4i gv4i;  // global
typedef __attribute__((address_space(3))) v4i lv4i;  // LDS
#endif

union BF16Frag { v16bf v; uint4 u[2]; };

// ---------------------------------------------------------------------------
// GEMM: C[rows,NCOLS] = A[rows,128](bf16) x Bt[NCOLS,128](bf16, pre-transposed)
// A tile staged in LDS once per block (async-to-LDS when available),
// f32 accumulate via v_wmma_f32_16x16x32_bf16. One 16x16 tile per wave.
// ---------------------------------------------------------------------------
template <int NCOLS>
__global__ __launch_bounds__(256) void gemm_bf16_wmma(
    const __bf16* __restrict__ A, const __bf16* __restrict__ Bt,
    float* __restrict__ C, int rows) {
  constexpr int K = 128;
  constexpr int CTILES = NCOLS / 16;        // col tiles per row strip
  constexpr int RPB    = 16 * (8 / CTILES); // rows per block (8 waves)
  __shared__ __align__(16) __bf16 As[RPB * K];

  const int tid  = threadIdx.x;
  const int wave = tid >> 5;
  const int lane = tid & 31;
  const int blockRow0 = blockIdx.x * RPB;

  // ---- stage A tile into LDS: 16B per thread per iteration -----------------
  for (int idx = tid * 8; idx < RPB * K; idx += 256 * 8) {
    const __bf16* g = A + (size_t)blockRow0 * K + idx;
#if HAVE_ASYNC_LDS
    __builtin_amdgcn_global_load_async_to_lds_b128(
        (gv4i*)(void*)g, (lv4i*)(void*)&As[idx], 0, 0);
#else
    *(uint4*)&As[idx] = *(const uint4*)g;
#endif
  }
#if HAVE_ASYNC_LDS
  __builtin_amdgcn_s_wait_asynccnt(0);
#endif
  __syncthreads();

  const int m0 = blockRow0 + (wave / CTILES) * 16;
  if (m0 >= rows) return;                   // wave-uniform: EXEC stays all-1s
  const int n0 = (wave % CTILES) * 16;
  const int hiHalf = lane >> 4;             // 0: lanes 0-15, 1: lanes 16-31

  // A-frag from LDS: row (lane&15) of this wave's 16-row tile
  const __bf16* arow = As + ((wave / CTILES) * 16 + (lane & 15)) * K +
                       (hiHalf ? 8 : 0);
  // B-frag from global: col n0+(lane&15); K at k0+(hiHalf?8:0)+{0..7,16..23}
  const __bf16* brow = Bt + (size_t)(n0 + (lane & 15)) * K + (hiHalf ? 8 : 0);

  v8f acc = {0.f, 0.f, 0.f, 0.f, 0.f, 0.f, 0.f, 0.f};
#pragma unroll
  for (int k0 = 0; k0 < K; k0 += 32) {
    BF16Frag a, b;
    a.u[0] = *(const uint4*)(arow + k0);
    a.u[1] = *(const uint4*)(arow + k0 + 16);
    b.u[0] = *(const uint4*)(brow + k0);
    b.u[1] = *(const uint4*)(brow + k0 + 16);
    acc = __builtin_amdgcn_wmma_f32_16x16x32_bf16(
        false, a.v, false, b.v, (short)0, acc, false, false);
  }
  const int crow0 = m0 + (hiHalf ? 8 : 0);
  const int ccol  = n0 + (lane & 15);
#pragma unroll
  for (int i = 0; i < 8; ++i)
    C[(size_t)(crow0 + i) * NCOLS + ccol] = acc[i];
}

// ---------------------------------------------------------------------------
// Small prep / utility kernels
// ---------------------------------------------------------------------------
__global__ void k_f32_to_bf16(const float* __restrict__ in,
                              __bf16* __restrict__ out, int n) {
  int i = blockIdx.x * blockDim.x + threadIdx.x;
  if (i < n) out[i] = (__bf16)in[i];
}

__global__ void k_relu_to_bf16(const float* __restrict__ in,
                               __bf16* __restrict__ out, int n) {
  int i = blockIdx.x * blockDim.x + threadIdx.x;
  if (i < n) out[i] = (__bf16)fmaxf(in[i], 0.0f);
}

// W[K][Ncol] -> Wt[Ncol][K] in bf16
__global__ void k_transpose_bf16(const float* __restrict__ W,
                                 __bf16* __restrict__ Wt, int K, int Ncol) {
  int i = blockIdx.x * blockDim.x + threadIdx.x;
  if (i >= K * Ncol) return;
  int k = i / Ncol, n = i - k * Ncol;
  Wt[n * K + k] = (__bf16)W[i];
}

__global__ void k_fill_f32(float* p, float v, int n) {
  int i = blockIdx.x * blockDim.x + threadIdx.x;
  if (i < n) p[i] = v;
}

__global__ void k_init_bias(float* __restrict__ out,
                            const float* __restrict__ bias, int rows, int cols) {
  int i = blockIdx.x * blockDim.x + threadIdx.x;
  if (i < rows * cols) out[i] = bias[i % cols];
}

// cvec[0..3] = sum_c We1[h*32+c]*ae1[h*32+c] ; cvec[4] = sum_c We2[c]*ae2[c]
__global__ void k_edge_coef(const float* We1, const float* ae1,
                            const float* We2, const float* ae2, float* cvec) {
  if (threadIdx.x == 0 && blockIdx.x == 0) {
    for (int h = 0; h < 4; ++h) {
      float s = 0.f;
      for (int c = 0; c < 32; ++c) s += We1[h * 32 + c] * ae1[h * 32 + c];
      cvec[h] = s;
    }
    float s2 = 0.f;
    for (int c = 0; c < 64; ++c) s2 += We2[c] * ae2[c];
    cvec[4] = s2;
  }
}

// ---------------------------------------------------------------------------
// a_src[n,h] = sum_c h[n,h*Cph+c]*as[...] (same for a_dst).
// One wave per node: lane loads HC/32 channels, butterfly-reduce per head.
// ---------------------------------------------------------------------------
template <int H, int HC>
__global__ __launch_bounds__(256) void k_attn_coef(
    const float* __restrict__ h, const float* __restrict__ as_,
    const float* __restrict__ ad_, float* __restrict__ a_src,
    float* __restrict__ a_dst) {
  constexpr int CPL = HC / 32;   // channels per lane
  constexpr int LPH = 32 / H;    // lanes per head
  const int n = blockIdx.x * 8 + (threadIdx.x >> 5);
  if (n >= NNODES) return;
  const int lane = threadIdx.x & 31;
  const int c0 = lane * CPL;
  const float* hp = h + (size_t)n * HC + c0;
  float ss = 0.f, sd = 0.f;
#pragma unroll
  for (int j = 0; j < CPL; ++j) {
    float v = hp[j];
    ss += v * as_[c0 + j];
    sd += v * ad_[c0 + j];
  }
#pragma unroll
  for (int m = LPH >> 1; m; m >>= 1) {
    ss += __shfl_xor(ss, m, 32);
    sd += __shfl_xor(sd, m, 32);
  }
  if ((lane & (LPH - 1)) == 0) {
    int hh = lane / LPH;
    a_src[n * H + hh] = ss;
    a_dst[n * H + hh] = sd;
  }
}

// ---------------------------------------------------------------------------
// Edge-phase kernels (3 streaming passes over E)
// ---------------------------------------------------------------------------
__device__ inline void atomicMaxF(float* addr, float val) {
  if (val >= 0.0f) atomicMax((int*)addr, __float_as_int(val));
  else             atomicMin((unsigned int*)addr, __float_as_uint(val));
}

// pass 1: alpha = leaky_relu(a_src[src]+a_dst[dst]+ea*c); segment-max into amax
__global__ void k_edge_alpha(const int* __restrict__ src, const int* __restrict__ dst,
                             const float* __restrict__ eattr,
                             const float* __restrict__ a_src,
                             const float* __restrict__ a_dst,
                             const float* __restrict__ cvec,
                             float* __restrict__ alpha, float* __restrict__ amax,
                             int H) {
  int t = blockIdx.x * blockDim.x + threadIdx.x;
  if (t >= NEDGES * H) return;
  int e = t / H, hh = t - e * H;
  int s = src[e], d = dst[e];
  float a = a_src[s * H + hh] + a_dst[d * H + hh] + eattr[e] * cvec[hh];
  a = (a > 0.f) ? a : NEG_SLOPE * a;
  alpha[t] = a;
  atomicMaxF(&amax[d * H + hh], a);
}

// pass 2: ex = exp(alpha - amax[dst]); segment-sum into denom; alpha <- ex
__global__ void k_edge_expsum(const int* __restrict__ dst,
                              float* __restrict__ alpha,
                              const float* __restrict__ amax,
                              float* __restrict__ denom, int H) {
  int t = blockIdx.x * blockDim.x + threadIdx.x;
  if (t >= NEDGES * H) return;
  int e = t / H, hh = t - e * H;
  int d = dst[e];
  float ex = expf(alpha[t] - amax[d * H + hh]);
  alpha[t] = ex;
  atomicAdd(&denom[d * H + hh], ex);
}

// pass 3: out[dst] += h[src] * (ex / (denom[dst]+1e-16)).
// One wave per edge (persistent, grid-stride) + prefetch of next source row.
template <int HC, int H>
__global__ __launch_bounds__(256) void k_edge_msg(
    const int* __restrict__ src, const int* __restrict__ dst,
    const float* __restrict__ alpha, const float* __restrict__ denom,
    const float* __restrict__ h, float* __restrict__ out) {
  constexpr int CPL = HC / 32;   // channels per lane (4 or 2)
  constexpr int Cph = HC / H;    // channels per head
  const int lane = threadIdx.x & 31;
  const int waveId = blockIdx.x * 8 + (threadIdx.x >> 5);
  const int stride = gridDim.x * 8;
  const int c0 = lane * CPL;
  const int hh = c0 / Cph;
  for (int e = waveId; e < NEDGES; e += stride) {
    int en = e + stride;
    if (en < NEDGES)  // hide gather latency of the next iteration
      __builtin_prefetch(h + (size_t)src[en] * HC + c0, 0, 1);
    int s = src[e], d = dst[e];
    float att = alpha[(size_t)e * H + hh] / (denom[(size_t)d * H + hh] + 1e-16f);
    const float* hp = h + (size_t)s * HC + c0;
    float* op = out + (size_t)d * HC + c0;
    if constexpr (CPL == 4) {
      const float4 v = *(const float4*)hp;
      atomicAdd(op + 0, v.x * att);
      atomicAdd(op + 1, v.y * att);
      atomicAdd(op + 2, v.z * att);
      atomicAdd(op + 3, v.w * att);
    } else {
      const float2 v = *(const float2*)hp;
      atomicAdd(op + 0, v.x * att);
      atomicAdd(op + 1, v.y * att);
    }
  }
}

// ---------------------------------------------------------------------------
// Pooling + prediction head
// ---------------------------------------------------------------------------
__global__ void k_pool(const float* __restrict__ out2, const int* __restrict__ batch,
                       float* __restrict__ sums, float* __restrict__ counts) {
  int t = blockIdx.x * blockDim.x + threadIdx.x;
  if (t >= NNODES * 16) return;
  int n = t >> 4, g = t & 15, c0 = g * 4;
  int b = batch[n];
  const float4 v = *(const float4*)(out2 + (size_t)n * 64 + c0);
  atomicAdd(&sums[b * 64 + c0 + 0], v.x);
  atomicAdd(&sums[b * 64 + c0 + 1], v.y);
  atomicAdd(&sums[b * 64 + c0 + 2], v.z);
  atomicAdd(&sums[b * 64 + c0 + 3], v.w);
  if (g == 0) atomicAdd(&counts[b], 1.0f);
}

__global__ void k_head(const float* __restrict__ sums, const float* __restrict__ counts,
                       const float* __restrict__ Wp, const float* __restrict__ bp,
                       float* __restrict__ out) {
  int g = threadIdx.x;
  if (g >= NGRAPH) return;
  float acc = 0.f;
  for (int c = 0; c < 64; ++c) acc += sums[g * 64 + c] * Wp[c];
  out[g] = acc / counts[g] + bp[0];
}

// ---------------------------------------------------------------------------
extern "C" void kernel_launch(void* const* d_in, const int* in_sizes, int n_in,
                              void* d_out, int out_size, void* d_ws, size_t ws_size,
                              hipStream_t stream) {
  const float* x     = (const float*)d_in[0];
  const int*   ei    = (const int*)  d_in[1];
  const float* eattr = (const float*)d_in[2];
  const int*   batch = (const int*)  d_in[3];
  const float* W1    = (const float*)d_in[4];
  const float* as1   = (const float*)d_in[5];
  const float* ad1   = (const float*)d_in[6];
  const float* We1   = (const float*)d_in[7];
  const float* ae1   = (const float*)d_in[8];
  const float* b1    = (const float*)d_in[9];
  const float* W2    = (const float*)d_in[10];
  const float* as2   = (const float*)d_in[11];
  const float* ad2   = (const float*)d_in[12];
  const float* We2   = (const float*)d_in[13];
  const float* ae2   = (const float*)d_in[14];
  const float* b2    = (const float*)d_in[15];
  const float* Wp    = (const float*)d_in[16];
  const float* bp    = (const float*)d_in[17];
  float* out = (float*)d_out;

  const int N = NNODES, E = NEDGES, G = NGRAPH;
  const int* src = ei;
  const int* dst = ei + E;

  // carve workspace (256B aligned)
  char* wsb = (char*)d_ws;
  size_t off = 0;
  auto alloc = [&](size_t bytes) -> void* {
    void* p = wsb + off; off += (bytes + 255) & ~(size_t)255; return p;
  };
  __bf16* xb    = (__bf16*)alloc((size_t)N * 128 * 2); // also reused for relu(h1)
  __bf16* W1t   = (__bf16*)alloc(128 * 128 * 2);
  __bf16* W2t   = (__bf16*)alloc(64 * 128 * 2);
  float*  h1    = (float*)alloc((size_t)N * 128 * 4);  // also reused as h2 [N,64]
  float*  out1  = (float*)alloc((size_t)N * 128 * 4);
  float*  asrc  = (float*)alloc((size_t)N * 4 * 4);
  float*  adst  = (float*)alloc((size_t)N * 4 * 4);
  float*  amax  = (float*)alloc((size_t)N * 4 * 4);
  float*  denom = (float*)alloc((size_t)N * 4 * 4);
  float*  alpha = (float*)alloc((size_t)E * 4 * 4);
  float*  out2  = (float*)alloc((size_t)N * 64 * 4);
  float*  sums  = (float*)alloc((size_t)G * 64 * 4);
  float*  cnts  = (float*)alloc((size_t)G * 4);
  float*  cvec  = (float*)alloc(8 * 4);

  const int T = 256;
  auto blk = [](long n, int t) { return (int)((n + t - 1) / t); };
  const int MSG_BLOCKS = 4096;  // persistent waves for the scatter pass

  // ---- prep: bf16 conversions, transposed weights, edge-attr coefficients
  k_f32_to_bf16<<<blk((long)N * 128, T), T, 0, stream>>>(x, xb, N * 128);
  k_transpose_bf16<<<blk(128 * 128, T), T, 0, stream>>>(W1, W1t, 128, 128);
  k_transpose_bf16<<<blk(128 * 64, T), T, 0, stream>>>(W2, W2t, 128, 64);
  k_edge_coef<<<1, 32, 0, stream>>>(We1, ae1, We2, ae2, cvec);

  // ---- layer 1: GEMM (WMMA), attention, 3 edge passes
  gemm_bf16_wmma<128><<<N / 16, 256, 0, stream>>>(xb, W1t, h1, N);
  k_attn_coef<4, 128><<<N / 8, 256, 0, stream>>>(h1, as1, ad1, asrc, adst);
  k_fill_f32<<<blk((long)N * 4, T), T, 0, stream>>>(amax, -INFINITY, N * 4);
  k_fill_f32<<<blk((long)N * 4, T), T, 0, stream>>>(denom, 0.0f, N * 4);
  k_init_bias<<<blk((long)N * 128, T), T, 0, stream>>>(out1, b1, N, 128);
  k_edge_alpha<<<blk((long)E * 4, T), T, 0, stream>>>(src, dst, eattr, asrc, adst,
                                                      cvec, alpha, amax, 4);
  k_edge_expsum<<<blk((long)E * 4, T), T, 0, stream>>>(dst, alpha, amax, denom, 4);
  k_edge_msg<128, 4><<<MSG_BLOCKS, 256, 0, stream>>>(src, dst, alpha, denom,
                                                     h1, out1);

  // ---- relu + bf16, layer 2
  k_relu_to_bf16<<<blk((long)N * 128, T), T, 0, stream>>>(out1, xb, N * 128);
  gemm_bf16_wmma<64><<<(N + 31) / 32, 256, 0, stream>>>(xb, W2t, h1, N); // h2 = h1
  k_attn_coef<1, 64><<<N / 8, 256, 0, stream>>>(h1, as2, ad2, asrc, adst);
  k_fill_f32<<<blk((long)N, T), T, 0, stream>>>(amax, -INFINITY, N);
  k_fill_f32<<<blk((long)N, T), T, 0, stream>>>(denom, 0.0f, N);
  k_init_bias<<<blk((long)N * 64, T), T, 0, stream>>>(out2, b2, N, 64);
  k_edge_alpha<<<blk((long)E, T), T, 0, stream>>>(src, dst, eattr, asrc, adst,
                                                  cvec + 4, alpha, amax, 1);
  k_edge_expsum<<<blk((long)E, T), T, 0, stream>>>(dst, alpha, amax, denom, 1);
  k_edge_msg<64, 1><<<MSG_BLOCKS, 256, 0, stream>>>(src, dst, alpha, denom,
                                                    h1, out2);

  // ---- pooling + head
  k_fill_f32<<<blk(G * 64, T), T, 0, stream>>>(sums, 0.0f, G * 64);
  k_fill_f32<<<1, G, 0, stream>>>(cnts, 0.0f, G);
  k_pool<<<blk((long)N * 16, T), T, 0, stream>>>(out2, batch, sums, cnts);
  k_head<<<1, 64, 0, stream>>>(sums, cnts, Wp, bp, out);
}
